// GNNEncoder_78718160601377
// MI455X (gfx1250) — compile-verified, matrix operand order checked by reading, not run on hardware
//
#include <hip/hip_runtime.h>
#include <hip/hip_bf16.h>

#define N_NODES 100000
#define N_EDGES 1600000
#define DIM 64
#define EDIM 32
#define NEG_SLOPE 0.2f

typedef __attribute__((ext_vector_type(16))) __bf16 v16bf;
typedef __attribute__((ext_vector_type(8)))  float  v8f;

__device__ __forceinline__ unsigned short f2bf(float f) {
    union { float f; unsigned u; } v; v.f = f;
    unsigned r = v.u + 0x7FFFu + ((v.u >> 16) & 1u);   // RNE
    return (unsigned short)(r >> 16);
}

// Pack row-major W[K][64] (f32) into the WMMA B-operand lane layout (bf16).
// Layout per 16x16x32 step: lane ln(0..15),hi: holds K = kblk*32 + hi*8 + {0..7}
// and +16..23 for column n = tile*16 + ln  (ISA 7.12.2, 16-bit B = transpose of A pattern).
__global__ void pack_w_kernel(const float* __restrict__ W, int K,
                              unsigned short* __restrict__ out) {
    int idx = blockIdx.x * blockDim.x + threadIdx.x;
    int total = K * 64;
    if (idx >= total) return;
    int j    = idx & 15;
    int lane = (idx >> 4) & 31;
    int tile = (idx >> 9) & 3;
    int kblk = idx >> 11;
    int hi = lane >> 4, ln = lane & 15;
    int k = kblk * 32 + hi * 8 + ((j >> 3) << 4) + (j & 7);
    int n = tile * 16 + ln;
    out[idx] = f2bf(W[k * 64 + n]);
}

// out[i][:] = bias ; m = -inf ; denom = 0
__global__ void init_kernel(float* __restrict__ out, const float* __restrict__ bias,
                            unsigned int* __restrict__ mbuf, float* __restrict__ denom) {
    int i = blockIdx.x * blockDim.x + threadIdx.x;
    if (i < N_NODES * DIM) out[i] = bias[i & (DIM - 1)];
    if (i < N_NODES) { mbuf[i] = 0xFF800000u; denom[i] = 0.0f; }
}

// h_src = relu?(x) @ W_src + b_src ; h_dst = relu?(x) @ W_dst + b_dst
// One wave per 16-row tile; 2 K-steps of v_wmma_f32_16x16x32_bf16 per 16x16 output tile.
__global__ void __launch_bounds__(256)
node_linear_kernel(const float* __restrict__ xin,
                   const unsigned short* __restrict__ wsrc_pk, const float* __restrict__ b_src,
                   const unsigned short* __restrict__ wdst_pk, const float* __restrict__ b_dst,
                   float* __restrict__ h_src, float* __restrict__ h_dst, int relu_in) {
    int wave = (blockIdx.x * blockDim.x + threadIdx.x) >> 5;
    int lane = threadIdx.x & 31;
    int m0 = wave * 16;
    if (m0 >= N_NODES) return;
    int hi = lane >> 4, ln = lane & 15;
    int row = m0 + ln;

    // A operand (16x32 bf16) per K-block: lane ln,hi holds row m0+ln,
    // K = kb*32 + hi*8 + {0..7} and {16..23}  (ISA 7.12.2)
    v16bf a[2];
#pragma unroll
    for (int kb = 0; kb < 2; ++kb) {
        const float* p = xin + row * DIM + kb * 32 + hi * 8;
#pragma unroll
        for (int t = 0; t < 8; ++t) {
            float f0 = p[t], f1 = p[16 + t];
            if (relu_in) { f0 = fmaxf(f0, 0.f); f1 = fmaxf(f1, 0.f); }
            a[kb][t]     = (__bf16)f0;
            a[kb][8 + t] = (__bf16)f1;
        }
    }

    const v16bf* Bs = (const v16bf*)wsrc_pk;
    const v16bf* Bd = (const v16bf*)wdst_pk;
#pragma unroll
    for (int tile = 0; tile < 4; ++tile) {
        int n = tile * 16 + ln;
        {
            float b = b_src[n];
            v8f acc;
#pragma unroll
            for (int r = 0; r < 8; ++r) acc[r] = b;
            acc = __builtin_amdgcn_wmma_f32_16x16x32_bf16(false, a[0], false,
                      Bs[(0 * 4 + tile) * 32 + lane], (short)0, acc, false, false);
            acc = __builtin_amdgcn_wmma_f32_16x16x32_bf16(false, a[1], false,
                      Bs[(1 * 4 + tile) * 32 + lane], (short)0, acc, false, false);
#pragma unroll
            for (int r = 0; r < 8; ++r) h_src[(m0 + hi * 8 + r) * DIM + n] = acc[r];
        }
        {
            float b = b_dst[n];
            v8f acc;
#pragma unroll
            for (int r = 0; r < 8; ++r) acc[r] = b;
            acc = __builtin_amdgcn_wmma_f32_16x16x32_bf16(false, a[0], false,
                      Bd[(0 * 4 + tile) * 32 + lane], (short)0, acc, false, false);
            acc = __builtin_amdgcn_wmma_f32_16x16x32_bf16(false, a[1], false,
                      Bd[(1 * 4 + tile) * 32 + lane], (short)0, acc, false, false);
#pragma unroll
            for (int r = 0; r < 8; ++r) h_dst[(m0 + hi * 8 + r) * DIM + n] = acc[r];
        }
    }
}

// Per-edge score: h_edge = edge_attr @ W_edge (WMMA, 16 edges/wave),
// v = h_edge + h_src[src] + h_dst[dst]; score = sum(leaky_relu(v) * att).
// Also atomic segment-max into mbuf (int/uint trick for float max).
__global__ void __launch_bounds__(256)
edge_score_kernel(const float* __restrict__ edge_attr,
                  const int* __restrict__ srcI, const int* __restrict__ dstI,
                  const unsigned short* __restrict__ wedge_pk,
                  const float* __restrict__ h_src, const float* __restrict__ h_dst,
                  const float* __restrict__ att,
                  float* __restrict__ score, unsigned int* __restrict__ mbuf) {
    int wave = (blockIdx.x * blockDim.x + threadIdx.x) >> 5;
    int lane = threadIdx.x & 31;
    int e0 = wave * 16;
    if (e0 >= N_EDGES) return;
    int hi = lane >> 4, ln = lane & 15;
    int row = e0 + ln;

    v16bf a;  // 16x32 A operand: edge_attr rows, K=EDIM=32 (single K-block)
    const float* p = edge_attr + row * EDIM + hi * 8;
#pragma unroll
    for (int t = 0; t < 8; ++t) { a[t] = (__bf16)p[t]; a[8 + t] = (__bf16)p[16 + t]; }

    int sI[8], dI[8];
#pragma unroll
    for (int r = 0; r < 8; ++r) {
        int m = e0 + hi * 8 + r;
        sI[r] = srcI[m]; dI[r] = dstI[m];
    }

    float s8[8];
#pragma unroll
    for (int r = 0; r < 8; ++r) s8[r] = 0.f;

    const v16bf* Be = (const v16bf*)wedge_pk;
#pragma unroll
    for (int tile = 0; tile < 4; ++tile) {
        int n = tile * 16 + ln;
        v8f acc;
#pragma unroll
        for (int r = 0; r < 8; ++r) acc[r] = 0.f;
        acc = __builtin_amdgcn_wmma_f32_16x16x32_bf16(false, a, false,
                  Be[tile * 32 + lane], (short)0, acc, false, false);
        float an = att[n];
#pragma unroll
        for (int r = 0; r < 8; ++r) {
            // D layout: VGPR r <-> edge m0+hi*8+r, lane ln <-> column n (coalesced gathers)
            float v = acc[r] + h_src[sI[r] * DIM + n] + h_dst[dI[r] * DIM + n];
            v = (v > 0.f) ? v : NEG_SLOPE * v;
            s8[r] += v * an;
        }
    }
    // reduce across the 16 lanes of each half-wave (xor masks < 16 stay in half)
#pragma unroll
    for (int r = 0; r < 8; ++r) {
        float s = s8[r];
        s += __shfl_xor(s, 8, 32);
        s += __shfl_xor(s, 4, 32);
        s += __shfl_xor(s, 2, 32);
        s += __shfl_xor(s, 1, 32);
        s8[r] = s;
    }
    if (ln == 0) {  // lanes 0 and 16 each own 8 edges
#pragma unroll
        for (int r = 0; r < 8; ++r) {
            int m = e0 + hi * 8 + r;
            float s = s8[r];
            score[m] = s;
            unsigned int* addr = mbuf + dI[r];
            if (s >= 0.f) atomicMax((int*)addr, __float_as_int(s));
            else          atomicMin(addr, __float_as_uint(s));
        }
    }
}

// ex = exp(score - m[dst]); denom[dst] += ex  (score overwritten with ex)
__global__ void edge_exp_kernel(const int* __restrict__ dstI,
                                const unsigned int* __restrict__ mbuf,
                                float* __restrict__ score, float* __restrict__ denom) {
    int e = blockIdx.x * blockDim.x + threadIdx.x;
    if (e >= N_EDGES) return;
    int d = dstI[e];
    float ex = __expf(score[e] - __uint_as_float(mbuf[d]));
    score[e] = ex;
    unsafeAtomicAdd(denom + d, ex);
}

// out[dst] += h_src[src] * (ex / denom[dst]) ; 4 threads per edge, 16 cols each
__global__ void edge_aggregate_kernel(const int* __restrict__ srcI, const int* __restrict__ dstI,
                                      const float* __restrict__ score, const float* __restrict__ denom,
                                      const float* __restrict__ h_src, float* __restrict__ out) {
    int t = blockIdx.x * blockDim.x + threadIdx.x;
    if (t >= N_EDGES * 4) return;
    int e  = t >> 2;
    int c0 = (t & 3) * 16;
    int s = srcI[e], d = dstI[e];
    float alpha = score[e] / denom[d];
    const float4* hs = (const float4*)(h_src + s * DIM + c0);
    float* op = out + d * DIM + c0;
#pragma unroll
    for (int q = 0; q < 4; ++q) {
        float4 v = hs[q];
        unsafeAtomicAdd(op + q * 4 + 0, v.x * alpha);
        unsafeAtomicAdd(op + q * 4 + 1, v.y * alpha);
        unsafeAtomicAdd(op + q * 4 + 2, v.z * alpha);
        unsafeAtomicAdd(op + q * 4 + 3, v.w * alpha);
    }
}

extern "C" void kernel_launch(void* const* d_in, const int* in_sizes, int n_in,
                              void* d_out, int out_size, void* d_ws, size_t ws_size,
                              hipStream_t stream) {
    const float* x   = (const float*)d_in[0];
    const int*   ei  = (const int*)d_in[1];
    const float* ea  = (const float*)d_in[2];
    const int* srcI = ei;
    const int* dstI = ei + N_EDGES;

    char* ws = (char*)d_ws;
    size_t o = 0;
    auto alloc = [&](size_t bytes) -> void* {
        void* p = ws + o;
        o += (bytes + 255) & ~(size_t)255;
        return p;
    };
    float*        hA = (float*)alloc((size_t)N_NODES * DIM * 4);
    float*        hB = (float*)alloc((size_t)N_NODES * DIM * 4);
    float*        hS = (float*)alloc((size_t)N_NODES * DIM * 4);
    float*        hD = (float*)alloc((size_t)N_NODES * DIM * 4);
    float*        sc = (float*)alloc((size_t)N_EDGES * 4);
    unsigned int* mb = (unsigned int*)alloc((size_t)N_NODES * 4);
    float*        dn = (float*)alloc((size_t)N_NODES * 4);
    unsigned short* wpk[3][3];
    for (int l = 0; l < 3; ++l) {
        wpk[l][0] = (unsigned short*)alloc(64 * 64 * 2);  // W_src packed
        wpk[l][1] = (unsigned short*)alloc(64 * 64 * 2);  // W_dst packed
        wpk[l][2] = (unsigned short*)alloc(32 * 64 * 2);  // W_edge packed
    }

    // Pack all weights into WMMA B-operand layout (tiny; runs once per launch)
    for (int l = 0; l < 3; ++l) {
        const float* Wsrc = (const float*)d_in[3 + l * 7 + 0];
        const float* Wdst = (const float*)d_in[3 + l * 7 + 2];
        const float* Wedg = (const float*)d_in[3 + l * 7 + 4];
        pack_w_kernel<<<(64 * 64 + 255) / 256, 256, 0, stream>>>(Wsrc, 64, wpk[l][0]);
        pack_w_kernel<<<(64 * 64 + 255) / 256, 256, 0, stream>>>(Wdst, 64, wpk[l][1]);
        pack_w_kernel<<<(32 * 64 + 255) / 256, 256, 0, stream>>>(Wedg, 32, wpk[l][2]);
    }

    const float* lin[3]  = { x, hA, hB };
    float*       lout[3] = { hA, hB, (float*)d_out };

    for (int l = 0; l < 3; ++l) {
        const float* b_src = (const float*)d_in[3 + l * 7 + 1];
        const float* b_dst = (const float*)d_in[3 + l * 7 + 3];
        const float* att   = (const float*)d_in[3 + l * 7 + 5];
        const float* bias  = (const float*)d_in[3 + l * 7 + 6];

        init_kernel<<<(N_NODES * DIM + 255) / 256, 256, 0, stream>>>(lout[l], bias, mb, dn);

        // 6250 waves (16 rows each), 8 waves/block
        node_linear_kernel<<<(6250 + 7) / 8, 256, 0, stream>>>(
            lin[l], wpk[l][0], b_src, wpk[l][1], b_dst, hS, hD, l > 0 ? 1 : 0);

        // 100000 waves (16 edges each), 8 waves/block
        edge_score_kernel<<<N_EDGES / 16 / 8, 256, 0, stream>>>(
            ea, srcI, dstI, wpk[l][2], hS, hD, att, sc, mb);

        edge_exp_kernel<<<(N_EDGES + 255) / 256, 256, 0, stream>>>(dstI, mb, sc, dn);

        edge_aggregate_kernel<<<(N_EDGES * 4 + 255) / 256, 256, 0, stream>>>(
            srcI, dstI, sc, dn, hS, lout[l]);
    }
}
